// IrrepConv1dBlock_83451214561662
// MI455X (gfx1250) — matrix operand order repaired and checked by reading, not compiled, not run on hardware
//
#include <hip/hip_runtime.h>

// ============================================================================
// IrrepConv1dBlock for MI455X (gfx1250, wave32, WMMA).
// All WMMA operand fragments are loaded as contiguous 16-byte chunks
// (2x ds_load_b128 / global_load_b128 per fragment) -- no per-element packing.
// ============================================================================

#define BATCH   128
#define CIN     256
#define COUT    256
#define NPTS    64
#define KW      5
#define DIR     16     // (LMAX+1)^2
#define NGRID   64     // RES*RES
#define NZ      (BATCH * NPTS)        // 8192
#define KCH     40                    // K chunks of 32 in Cin*KW = 1280

typedef __bf16 v16bf __attribute__((ext_vector_type(16)));
typedef float  v8f   __attribute__((ext_vector_type(8)));

union frag_u { v16bf v; unsigned short u[16]; };
struct __align__(16) h8 { unsigned short s[8]; };
union pk8 { h8 h; unsigned d[4]; };

__device__ __forceinline__ unsigned short f32_to_bf16(float f) {
    union { float f; unsigned u; } x; x.f = f;
    unsigned r = x.u + 0x7FFFu + ((x.u >> 16) & 1u);   // round-nearest-even
    return (unsigned short)(r >> 16);
}
__device__ __forceinline__ unsigned bpack(float lo, float hi) {
    return (unsigned)f32_to_bf16(lo) | ((unsigned)f32_to_bf16(hi) << 16);
}

// 16 contiguous bf16 -> fragment (two 16B loads)
__device__ __forceinline__ frag_u load_frag16(const unsigned short* p) {
    frag_u f;
    *(h8*)(&f.u[0]) = *(const h8*)(p);
    *(h8*)(&f.u[8]) = *(const h8*)(p + 8);
    return f;
}
// two separate contiguous 8-element runs -> fragment
__device__ __forceinline__ frag_u load_frag_2x8(const unsigned short* p0,
                                                const unsigned short* p1) {
    frag_u f;
    *(h8*)(&f.u[0]) = *(const h8*)(p0);
    *(h8*)(&f.u[8]) = *(const h8*)(p1);
    return f;
}

__device__ __forceinline__ v8f wmma_bf16(const frag_u& a, const frag_u& b, v8f c) {
    return __builtin_amdgcn_wmma_f32_16x16x32_bf16(false, a.v, false, b.v,
                                                   (short)0, c, false, false);
}

__device__ __forceinline__ int deg_of(int i) {
    return (i == 0) ? 0 : (i < 4) ? 1 : (i < 9) ? 2 : 3;
}
__device__ __forceinline__ float silu(float x) { return x / (1.0f + expf(-x)); }

// A-fragment element j -> K index (contiguous: k = 16*half + j)
__device__ __forceinline__ int ka_of(int j, int half) {
    return j + half * 16;
}

// ---------------------------------------------------------------------------
// Kernel 1: pack conv weights into WMMA A-fragment order (bf16).
// wfrag[l][chunk(40)][mtile(16)][lane(32)][j(16)], chunk = kk*8 + cinblk.
// ---------------------------------------------------------------------------
__global__ __launch_bounds__(256) void pack_wfrag(const float* __restrict__ w,
                                                  unsigned short* __restrict__ wfrag) {
    int idx = blockIdx.x * 256 + threadIdx.x;
    if (idx >= 4 * KCH * 16 * 32 * 16) return;
    int j     = idx & 15;
    int lane  = (idx >> 4) & 31;
    int mtile = (idx >> 9) & 15;
    int rest  = idx >> 13;             // l*40 + chunk
    int chunk = rest % KCH;
    int l     = rest / KCH;
    int kk = chunk >> 3, cb = chunk & 7;
    int half = lane >> 4, mr = lane & 15;
    int cout = mtile * 16 + mr;
    int cin  = cb * 32 + ka_of(j, half);
    wfrag[idx] = f32_to_bf16(w[((size_t)(l * COUT + cout) * CIN + cin) * KW + kk]);
}

// ---------------------------------------------------------------------------
// Kernel 2: gating_w -> bf16, SAME layout gwb[c][k] (fragment-contiguous in k)
// ---------------------------------------------------------------------------
__global__ __launch_bounds__(256) void pack_gw(const float* __restrict__ gw,
                                               unsigned short* __restrict__ gwb) {
    int idx = blockIdx.x * 256 + threadIdx.x;      // 65536
    gwb[idx] = f32_to_bf16(gw[idx]);
}

// ---------------------------------------------------------------------------
// Kernel 3: prebuild A fragments for to_grid (64x16, K padded to 32) and
// from_grid^T (16x64, 2 K-chunks).
// ---------------------------------------------------------------------------
__global__ __launch_bounds__(256) void pack_tgfg(const float* __restrict__ tg,
                                                 const float* __restrict__ fg,
                                                 unsigned short* __restrict__ tgf,
                                                 unsigned short* __restrict__ fgf) {
    int idx = blockIdx.x * 256 + threadIdx.x;
    if (idx < 4 * 32 * 16) {                       // tgfrag[mtile(4)][lane][j]
        int j = idx & 15, lane = (idx >> 4) & 31, mtile = idx >> 9;
        int half = lane >> 4, mr = lane & 15;
        int k = ka_of(j, half);
        int g = mtile * 16 + mr;
        tgf[idx] = (k < DIR) ? f32_to_bf16(tg[g * DIR + k]) : (unsigned short)0;
    } else if (idx < 4 * 32 * 16 + 2 * 32 * 16) {  // fgfrag[chunk(2)][lane][j]
        int id = idx - 4 * 32 * 16;
        int j = id & 15, lane = (id >> 4) & 31, chunk = id >> 9;
        int half = lane >> 4, mr = lane & 15;
        int g = chunk * 32 + ka_of(j, half);       // K index = grid point
        fgf[id] = f32_to_bf16(fg[g * DIR + mr]);   // row i = mr
    }
}

// ---------------------------------------------------------------------------
// Kernel 4: conv GEMM. LDS holds the x slice TRANSPOSED: xsw[n'][cin] so
// B fragments are 16 contiguous bf16 (2x ds_load_b128). Row stride 264
// halfwords = 528 B (16B multiple; 132 dwords == 4 mod 64 -> bank rotation).
// ---------------------------------------------------------------------------
__global__ __launch_bounds__(256) void conv_gemm(const float* __restrict__ x,
                                                 const unsigned short* __restrict__ wfrag,
                                                 const float* __restrict__ bias,
                                                 float* __restrict__ h) {
    const int bc = blockIdx.x;
    const int b = bc >> 4, comp = bc & 15;
    const int l = deg_of(comp);
    const int tid = threadIdx.x;

    __shared__ __align__(16) unsigned short xsw[68][264];  // [n+2][cin], pad rows 0,1,66,67

    for (int i = tid; i < 68 * 264 / 2; i += 256) ((unsigned int*)xsw)[i] = 0u;
    __syncthreads();
    {   // stage: thread handles cin pair (2 rows of x), half the n range; dword stores
        const int cpair = tid & 127;
        const int nh    = tid >> 7;
        const float4* x0 = (const float4*)(x + ((size_t)(b * CIN + cpair * 2 + 0) * DIR + comp) * NPTS);
        const float4* x1 = (const float4*)(x + ((size_t)(b * CIN + cpair * 2 + 1) * DIR + comp) * NPTS);
        #pragma unroll
        for (int q = 0; q < 8; ++q) {
            float4 a = x0[nh * 8 + q];
            float4 c = x1[nh * 8 + q];
            int n = nh * 32 + q * 4;
            *(unsigned*)&xsw[n + 2][cpair * 2] = bpack(a.x, c.x);
            *(unsigned*)&xsw[n + 3][cpair * 2] = bpack(a.y, c.y);
            *(unsigned*)&xsw[n + 4][cpair * 2] = bpack(a.z, c.z);
            *(unsigned*)&xsw[n + 5][cpair * 2] = bpack(a.w, c.w);
        }
    }
    __syncthreads();

    const int wave = tid >> 5, lane = tid & 31;
    const int half = lane >> 4, mr = lane & 15;
    const frag_u* wbase = (const frag_u*)wfrag + (size_t)l * KCH * 16 * 32;

    v8f acc[2][4] = {};
    for (int chunk = 0; chunk < KCH; ++chunk) {
        const int kk = chunk >> 3, cb = chunk & 7;
        frag_u a0 = wbase[(chunk * 16 + wave * 2 + 0) * 32 + lane];
        frag_u a1 = wbase[(chunk * 16 + wave * 2 + 1) * 32 + lane];
        if (chunk + 1 < KCH)
            __builtin_prefetch(&wbase[((chunk + 1) * 16 + wave * 2) * 32 + lane], 0, 1);
        frag_u bf[4];
        #pragma unroll
        for (int ni = 0; ni < 4; ++ni)
            bf[ni] = load_frag16(&xsw[ni * 16 + mr + kk][cb * 32 + half * 16]);
        #pragma unroll
        for (int ni = 0; ni < 4; ++ni) {
            acc[0][ni] = wmma_bf16(a0, bf[ni], acc[0][ni]);
            acc[1][ni] = wmma_bf16(a1, bf[ni], acc[1][ni]);
        }
    }

    #pragma unroll
    for (int mi = 0; mi < 2; ++mi) {
        const int cb0 = (wave * 2 + mi) * 16 + half * 8;   // 8 consecutive couts
        #pragma unroll
        for (int ni = 0; ni < 4; ++ni) {
            const int n = ni * 16 + mr;
            float* hp = h + ((size_t)(b * NPTS + n) * DIR + comp) * COUT + cb0;
            v8f t = acc[mi][ni];
            #pragma unroll
            for (int r = 0; r < 8; ++r) {
                float v = t[r];
                if (comp == 0) v += bias[cb0 + r];
                hp[r] = v;
            }
        }
    }
}

// ---------------------------------------------------------------------------
// Kernel 5: equivariant norm. One workgroup per z; thread owns column c.
// ---------------------------------------------------------------------------
__global__ __launch_bounds__(256) void norm_kernel(const float* __restrict__ h,
                                                   const float* __restrict__ affine,
                                                   unsigned short* __restrict__ hN) {
    const int z = blockIdx.x, c = threadIdx.x;
    const float* hz = h + (size_t)z * DIR * COUT;
    __shared__ float red[256];

    float vals[DIR];
    #pragma unroll
    for (int i = 0; i < DIR; ++i) vals[i] = hz[i * COUT + c];

    red[c] = vals[0];
    __syncthreads();
    for (int s = 128; s > 0; s >>= 1) { if (c < s) red[c] += red[c + s]; __syncthreads(); }
    const float mean0 = red[0] * (1.0f / 256.0f);
    __syncthreads();
    vals[0] -= mean0;

    float var = 0.0f;
    #pragma unroll
    for (int i = 0; i < DIR; ++i) {
        const int l = deg_of(i);
        const float bal = 1.0f / (float)((2 * l + 1) * 4);
        var += bal * vals[i] * vals[i];
    }
    red[c] = var;
    __syncthreads();
    for (int s = 128; s > 0; s >>= 1) { if (c < s) red[c] += red[c + s]; __syncthreads(); }
    const float inv = rsqrtf(red[0] * (1.0f / 256.0f) + 1e-5f);

    unsigned short* o = hN + (size_t)z * DIR * COUT;
    #pragma unroll
    for (int i = 0; i < DIR; ++i) {
        const float sc = inv * affine[deg_of(i) * COUT + c];
        o[i * COUT + c] = f32_to_bf16(vals[i] * sc);
    }
}

// ---------------------------------------------------------------------------
// Kernel 6: gating GEMM + SiLU -> out component 0.
// A fragments: 2x ds_load_b128 from lh[z][k]; B fragments: 2x global b128
// from gwb[c][k] (L2 resident).
// ---------------------------------------------------------------------------
__global__ __launch_bounds__(256) void gate_gemm(const unsigned short* __restrict__ hN,
                                                 const unsigned short* __restrict__ gwb,
                                                 const float* __restrict__ gb,
                                                 float* __restrict__ out) {
    const int zbase = blockIdx.x * 64;
    __shared__ __align__(16) unsigned short lh[64][264];

    for (int idx = threadIdx.x; idx < 64 * 256; idx += 256)
        lh[idx >> 8][idx & 255] = hN[(size_t)(zbase + (idx >> 8)) * DIR * COUT + (idx & 255)];
    __syncthreads();

    const int tid = threadIdx.x;
    const int wave = tid >> 5, lane = tid & 31;
    const int half = lane >> 4, mr = lane & 15;
    const int mtile = wave >> 1, nb = (wave & 1) * 8;

    const unsigned short* arow = &lh[mtile * 16 + mr][0];

    v8f acc[8] = {};
    for (int chunk = 0; chunk < 8; ++chunk) {
        frag_u a = load_frag_2x8(arow + chunk * 32 + half * 16,
                                 arow + chunk * 32 + half * 16 + 8);
        #pragma unroll
        for (int ni = 0; ni < 8; ++ni) {
            const int cc = (nb + ni) * 16 + mr;
            frag_u bf = load_frag16(gwb + (size_t)cc * 256 + chunk * 32 + half * 16);
            acc[ni] = wmma_bf16(a, bf, acc[ni]);
        }
    }

    #pragma unroll
    for (int ni = 0; ni < 8; ++ni) {
        const int c = (nb + ni) * 16 + mr;
        v8f t = acc[ni];
        #pragma unroll
        for (int r = 0; r < 8; ++r) {
            const int zg = zbase + mtile * 16 + r + half * 8;
            const int b = zg >> 6, n = zg & 63;
            out[((size_t)(b * COUT + c) * DIR + 0) * NPTS + n] = silu(t[r] + gb[c]);
        }
    }
}

// ---------------------------------------------------------------------------
// Kernel 7: per-z S2 grid round trip. Both LDS stages are kept transposed
// (index-major per channel) so B fragments are contiguous 16B loads, and
// GEMM1 results are written with one packed 16B ds_store per accumulator.
// ---------------------------------------------------------------------------
__global__ __launch_bounds__(256) void grid_kernel(const unsigned short* __restrict__ hN,
                                                   const unsigned short* __restrict__ tgf,
                                                   const unsigned short* __restrict__ fgf,
                                                   float* __restrict__ out) {
    const int z = blockIdx.x;
    __shared__ __align__(16) unsigned short lhn_t[256][24];    // [c][i]  (i: 0..15)
    __shared__ __align__(16) unsigned short lgrid_t[256][72];  // [c][g]  (g: 0..63)

    for (int idx = threadIdx.x; idx < DIR * 256; idx += 256)
        lhn_t[idx & 255][idx >> 8] = hN[(size_t)z * DIR * COUT + idx];
    __syncthreads();

    const int tid = threadIdx.x;
    const int wave = tid >> 5, lane = tid & 31;
    const int half = lane >> 4, mr = lane & 15;

    // GEMM1: grid[g][c] = silu(sum_i to_grid[g][i] * hN[i][c]); K=16 padded to 32
    {
        const int mtile = wave >> 1, nb = (wave & 1) * 8;
        frag_u a = ((const frag_u*)tgf)[mtile * 32 + lane];
        v8f acc[8] = {};
        #pragma unroll
        for (int ni = 0; ni < 8; ++ni) {
            const int c = (nb + ni) * 16 + mr;
            frag_u bf;
            if (half == 0) {
                bf = load_frag16(&lhn_t[c][0]);
            } else {
                #pragma unroll
                for (int j = 0; j < 16; ++j) bf.u[j] = 0;     // K >= 16 zero pad
            }
            acc[ni] = wmma_bf16(a, bf, acc[ni]);
        }
        #pragma unroll
        for (int ni = 0; ni < 8; ++ni) {
            const int c = (nb + ni) * 16 + mr;
            v8f t = acc[ni];
            pk8 o;
            #pragma unroll
            for (int r = 0; r < 4; ++r)
                o.d[r] = bpack(silu(t[2 * r]), silu(t[2 * r + 1]));
            *(h8*)&lgrid_t[c][mtile * 16 + half * 8] = o.h;   // 8 consecutive g
        }
    }
    __syncthreads();

    // GEMM2: s2[i][c] = sum_g from_grid[g][i] * grid[g][c]; K=64 (2 chunks)
    v8f acc2[2] = {};
    for (int chunk = 0; chunk < 2; ++chunk) {
        frag_u a = ((const frag_u*)fgf)[chunk * 32 + lane];
        #pragma unroll
        for (int t2 = 0; t2 < 2; ++t2) {
            const int c = (wave * 2 + t2) * 16 + mr;
            frag_u bf = load_frag16(&lgrid_t[c][chunk * 32 + half * 16]);
            acc2[t2] = wmma_bf16(a, bf, acc2[t2]);
        }
    }

    const int b = z >> 6, n = z & 63;
    #pragma unroll
    for (int t2 = 0; t2 < 2; ++t2) {
        const int c = (wave * 2 + t2) * 16 + mr;
        v8f t = acc2[t2];
        #pragma unroll
        for (int r = 0; r < 8; ++r) {
            const int i = r + half * 8;
            if (i > 0) out[((size_t)(b * COUT + c) * DIR + i) * NPTS + n] = t[r];
        }
    }
}

// ---------------------------------------------------------------------------
// Workspace layout (bytes): assumes ws_size >= ~210 MiB.
// ---------------------------------------------------------------------------
static const size_t WFRAG_OFF = 0;                        // 2.62 MB bf16 frags
static const size_t GWB_OFF   = (size_t)4  << 20;         // 128 KB
static const size_t TGF_OFF   = (size_t)5  << 20;         // 4 KB
static const size_t FGF_OFF   = ((size_t)5 << 20) + (64 << 10);
static const size_t H_OFF     = (size_t)8  << 20;         // 128 MiB fp32 h
static const size_t HN_OFF    = ((size_t)8 << 20) + ((size_t)128 << 20); // 64 MiB bf16

extern "C" void kernel_launch(void* const* d_in, const int* in_sizes, int n_in,
                              void* d_out, int out_size, void* d_ws, size_t ws_size,
                              hipStream_t stream) {
    const float* x    = (const float*)d_in[0];
    const float* w    = (const float*)d_in[1];
    const float* bias = (const float*)d_in[2];
    const float* aff  = (const float*)d_in[3];
    const float* gw   = (const float*)d_in[4];
    const float* gb   = (const float*)d_in[5];
    const float* tg   = (const float*)d_in[6];
    const float* fg   = (const float*)d_in[7];
    float* out = (float*)d_out;
    char* ws = (char*)d_ws;

    unsigned short* wfrag = (unsigned short*)(ws + WFRAG_OFF);
    unsigned short* gwb   = (unsigned short*)(ws + GWB_OFF);
    unsigned short* tgf   = (unsigned short*)(ws + TGF_OFF);
    unsigned short* fgf   = (unsigned short*)(ws + FGF_OFF);
    float*          h     = (float*)(ws + H_OFF);
    unsigned short* hN    = (unsigned short*)(ws + HN_OFF);

    pack_wfrag <<<(4 * KCH * 16 * 32 * 16) / 256, 256, 0, stream>>>(w, wfrag);
    pack_gw    <<<256, 256, 0, stream>>>(gw, gwb);
    pack_tgfg  <<<12, 256, 0, stream>>>(tg, fg, tgf, fgf);
    conv_gemm  <<<BATCH * DIR, 256, 0, stream>>>(x, wfrag, bias, h);
    norm_kernel<<<NZ, 256, 0, stream>>>(h, aff, hN);
    gate_gemm  <<<NZ / 64, 256, 0, stream>>>(hN, gwb, gb, out);
    grid_kernel<<<NZ, 256, 0, stream>>>(hN, tgf, fgf, out);
}
